// MultiHeadAttention_43696997269537
// MI455X (gfx1250) — compile-verified, matrix operand order checked by reading, not run on hardware
//
#include <hip/hip_runtime.h>
#include <hip/hip_bf16.h>

// Problem constants
#define TT   2048            // sequence length
#define CC   1024            // embed dim
#define HH   16              // heads
#define DD   64              // head dim
#define BB   2               // batch
#define MROWS 4096           // B*T

typedef __attribute__((ext_vector_type(16))) _Float16 v16h;
typedef __attribute__((ext_vector_type(8)))  _Float16 v8h;
typedef __attribute__((ext_vector_type(4)))  _Float16 v4h;
typedef __attribute__((ext_vector_type(8)))  float    v8f;
typedef __attribute__((ext_vector_type(4)))  float    v4f;

#define WMMA_F16(a,b,c) __builtin_amdgcn_wmma_f32_16x16x32_f16(false,(a),false,(b),(short)0,(c),false,false)

// ---------------------------------------------------------------------------
// CDNA5 async copy: 16 bytes global -> LDS per lane, tracked by ASYNCcnt.
// (ISA 10 / 15.18.3: GLOBAL_LOAD_ASYNC_TO_LDS_B128, GV addressing mode)
// ---------------------------------------------------------------------------
__device__ __forceinline__ void async_copy_b128(void* lds_dst, const void* gsrc) {
  auto l = (__attribute__((address_space(3))) void*)lds_dst;
  asm volatile("global_load_async_to_lds_b128 %0, %1, off"
               :: "v"(l), "v"(gsrc) : "memory");
}
__device__ __forceinline__ void wait_async0() {
  asm volatile("s_wait_asynccnt 0x0" ::: "memory");
}

// Load a 16x32 f16 fragment (row-major tile in LDS, leading dim `ld` halves,
// ld % 8 == 0, base 16B aligned). Per ISA 7.12.2 A/B-operand layout.
__device__ __forceinline__ v16h frag_ld(const _Float16* p, int ld) {
  const int lane = threadIdx.x & 31;
  const _Float16* q = p + (lane & 15) * ld + ((lane >> 4) << 3);
  v8h lo = *(const v8h*)(q);
  v8h hi = *(const v8h*)(q + 16);
  v16h out;
#pragma unroll
  for (int i = 0; i < 8; ++i) { out[i] = lo[i]; out[i + 8] = hi[i]; }
  return out;
}

__device__ __forceinline__ void atomicMaxFloatShared(float* addr, float val) {
  int* ia = (int*)addr;
  int old = *ia;
  while (__int_as_float(old) < val) {
    int assumed = old;
    old = atomicCAS(ia, assumed, __float_as_int(val));
    if (old == assumed) break;
  }
}

// ---------------------------------------------------------------------------
// Generic GEMM: out[M=4096, N=1024] = A[4096,1024] @ W[1024,1024] + bias
// A_MODE:   0 = A plain row-major fp32, 1 = A is y_head fp32 [B,H,T,D]
// OUT_MODE: 0 = f16 [B,H,T,D] ws, 1 = fp32 plain [M,N], 2 = f16 [B,H,D,T] ws
// Block tile 64x128, BK=32, 8 waves (2x4), wave tile 32x32 (2x2 WMMA frags).
// ---------------------------------------------------------------------------
template <int A_MODE, int OUT_MODE>
__global__ __launch_bounds__(256)
void mha_gemm_kernel(const float* __restrict__ A, const float* __restrict__ W,
                     const float* __restrict__ bias,
                     _Float16* __restrict__ outH, float* __restrict__ outF)
{
  __shared__ __attribute__((aligned(16))) _Float16 As[64 * 40];
  __shared__ __attribute__((aligned(16))) _Float16 Bt[128 * 40];   // transposed W tile
  const int tid  = threadIdx.x;
  const int wave = tid >> 5;
  const int m0 = blockIdx.y * 64;
  const int n0 = blockIdx.x * 128;
  const int wm = (wave >> 2) * 32;   // 0 or 32
  const int wn = (wave & 3) * 32;    // 0..96

  v8f acc[2][2];
  acc[0][0] = {}; acc[0][1] = {}; acc[1][0] = {}; acc[1][1] = {};

  for (int kt = 0; kt < CC; kt += 32) {
    __syncthreads();
    // Stage A 64x32 fp32 -> f16: one float4 load + one b64 LDS store per chunk
#pragma unroll
    for (int it = 0; it < 2; ++it) {
      int idx = tid + it * 256;          // 512 chunks of 4
      int r  = idx >> 3;                 // 0..63
      int c4 = idx & 7;                  // 0..7  (k chunk of 4)
      int k = kt + c4 * 4;
      const float* src;
      if (A_MODE == 0) {
        src = A + (size_t)(m0 + r) * CC + k;
      } else {
        int b = (m0 + r) >> 11, t = (m0 + r) & (TT - 1);
        int h = k >> 6, d = k & (DD - 1);
        src = A + ((size_t)(b * HH + h) * TT + t) * DD + d;   // 4 contiguous d
      }
      v4f f = *(const v4f*)src;
      v4h h4;
#pragma unroll
      for (int j = 0; j < 4; ++j) h4[j] = (_Float16)f[j];
      *(v4h*)&As[r * 40 + c4 * 4] = h4;
    }
    // Stage W tile transposed: Bt[n][k] = W[kt+k][n0+n]; pack 4 k per b64 store
#pragma unroll
    for (int it = 0; it < 4; ++it) {
      int idx = tid + it * 256;          // 1024 chunks
      int n  = idx & 127;
      int kq = idx >> 7;                 // 0..7 (k group of 4)
      v4h h4;
#pragma unroll
      for (int j = 0; j < 4; ++j)
        h4[j] = (_Float16)W[(size_t)(kt + kq * 4 + j) * CC + n0 + n];
      *(v4h*)&Bt[n * 40 + kq * 4] = h4;
    }
    __syncthreads();

    v16h a0 = frag_ld(&As[wm * 40], 40);
    v16h a1 = frag_ld(&As[(wm + 16) * 40], 40);
    v16h b0 = frag_ld(&Bt[wn * 40], 40);
    v16h b1 = frag_ld(&Bt[(wn + 16) * 40], 40);
    acc[0][0] = WMMA_F16(a0, b0, acc[0][0]);
    acc[0][1] = WMMA_F16(a0, b1, acc[0][1]);
    acc[1][0] = WMMA_F16(a1, b0, acc[1][0]);
    acc[1][1] = WMMA_F16(a1, b1, acc[1][1]);
  }

  const int lane = tid & 31;
  const int nl = lane & 15, g = lane >> 4;
#pragma unroll
  for (int sm = 0; sm < 2; ++sm)
#pragma unroll
    for (int sn = 0; sn < 2; ++sn)
#pragma unroll
      for (int r = 0; r < 8; ++r) {
        int row = m0 + wm + sm * 16 + r + g * 8;
        int col = n0 + wn + sn * 16 + nl;
        float v = acc[sm][sn][r] + bias[col];
        int b = row >> 11, t = row & (TT - 1);
        int h = col >> 6,  d = col & (DD - 1);
        if (OUT_MODE == 0) {
          outH[((size_t)(b * HH + h) * TT + t) * DD + d] = (_Float16)v;
        } else if (OUT_MODE == 2) {   // V stored pre-transposed [B,H,D,T]
          outH[((size_t)(b * HH + h) * DD + d) * TT + t] = (_Float16)v;
        } else {
          outF[(size_t)row * CC + col] = v;
        }
      }
}

// ---------------------------------------------------------------------------
// Scores + causal softmax. Grid (B*H, T/64). Block owns 64 query rows.
// Q/K tiles staged with GLOBAL_LOAD_ASYNC_TO_LDS_B128 (pure f16 copies).
// Raw score tiles bounce through LDS so global writes are coalesced float4s.
// ---------------------------------------------------------------------------
__global__ __launch_bounds__(256)
void mha_scores_kernel(const _Float16* __restrict__ qh,
                       const _Float16* __restrict__ kh,
                       float* __restrict__ att)
{
  __shared__ __attribute__((aligned(16))) _Float16 Qs[64 * 72];
  __shared__ __attribute__((aligned(16))) _Float16 Ks[64 * 72];
  __shared__ __attribute__((aligned(16))) float    Ss[64 * 68];   // score bounce tile
  __shared__ float rowmax[64];
  __shared__ float rowsum[64];
  const int tid  = threadIdx.x;
  const int lane = tid & 31;
  const int wave = tid >> 5;
  const int bh = blockIdx.x;
  const int qb = blockIdx.y;
  const int q0 = qb * 64;
  float* attB = att + (size_t)bh * TT * TT;

  if (tid < 64) { rowmax[tid] = -3.0e38f; rowsum[tid] = 0.0f; }
  // Q tile 64x64 halves: 512 x 16B async copies (2 per thread)
#pragma unroll
  for (int it = 0; it < 2; ++it) {
    int idx = tid + it * 256;
    int r = idx >> 2, c8 = idx & 3;          // row, 8-half chunk
    async_copy_b128(&Qs[r * 72 + c8 * 8],
                    qh + ((size_t)bh * TT + q0 + r) * DD + c8 * 8);
  }

  const int wm = (wave >> 1) * 16;   // 0..48
  const int wn = (wave & 1) * 32;    // 0 or 32
  const int nl = lane & 15, g = lane >> 4;
  float vmax[8];
#pragma unroll
  for (int r = 0; r < 8; ++r) vmax[r] = -3.0e38f;

  for (int jb = 0; jb <= qb; ++jb) {      // causal: key blocks up to diagonal
    __syncthreads();                      // prev Ss copy-out & Ks reads done
#pragma unroll
    for (int it = 0; it < 2; ++it) {
      int idx = tid + it * 256;
      int r = idx >> 2, c8 = idx & 3;
      async_copy_b128(&Ks[r * 72 + c8 * 8],
                      kh + ((size_t)bh * TT + jb * 64 + r) * DD + c8 * 8);
    }
    wait_async0();
    __syncthreads();
    // compute 64x64 score tile, track raw row max, stage into Ss
#pragma unroll
    for (int sn = 0; sn < 2; ++sn) {
      v8f acc = {};
#pragma unroll
      for (int kk = 0; kk < 2; ++kk) {   // D=64 -> two K=32 steps
        v16h a = frag_ld(&Qs[wm * 72 + kk * 32], 72);
        v16h b = frag_ld(&Ks[(wn + sn * 16) * 72 + kk * 32], 72);
        acc = WMMA_F16(a, b, acc);
      }
#pragma unroll
      for (int r = 0; r < 8; ++r) {
        vmax[r] = fmaxf(vmax[r], acc[r]);   // m >= true max is safe for softmax
        Ss[(wm + r + g * 8) * 68 + wn + sn * 16 + nl] = acc[r];
      }
    }
    __syncthreads();
    // coalesced masked writeback: 1024 float4 chunks, 4 per thread
#pragma unroll
    for (int it = 0; it < 4; ++it) {
      int idx = tid + it * 256;
      int r  = idx >> 4;                 // 0..63
      int c4 = idx & 15;                 // float4 chunk within 64
      int grow = q0 + r;
      int gcol = jb * 64 + c4 * 4;
      v4f s = *(const v4f*)&Ss[r * 68 + c4 * 4];
      v4f o;
#pragma unroll
      for (int j = 0; j < 4; ++j)
        o[j] = (gcol + j <= grow) ? s[j] * 0.125f : -1.0e30f;   // 1/sqrt(64)
      *(v4f*)(attB + (size_t)grow * TT + gcol) = o;
    }
  }
  // reduce row max across the 16-lane half holding the same rows
#pragma unroll
  for (int r = 0; r < 8; ++r) {
    float v = vmax[r] * 0.125f;
    v = fmaxf(v, __shfl_xor(v, 1, 32));
    v = fmaxf(v, __shfl_xor(v, 2, 32));
    v = fmaxf(v, __shfl_xor(v, 4, 32));
    v = fmaxf(v, __shfl_xor(v, 8, 32));
    if (nl == 0) atomicMaxFloatShared(&rowmax[wm + r + g * 8], v);
  }
  __syncthreads();

  const int ncols = q0 + 64;           // multiple of 64
  // Phase 2: e = exp(s - max), accumulate row sums (float4 vectorized)
  for (int row = 0; row < 64; ++row) {
    float m = rowmax[row];
    float lsum = 0.0f;
    float* rp = attB + (size_t)(q0 + row) * TT;
    for (int c = tid; c < (ncols >> 2); c += 256) {
      v4f x = *(v4f*)(rp + c * 4);
      v4f e;
#pragma unroll
      for (int j = 0; j < 4; ++j) { e[j] = __expf(x[j] - m); lsum += e[j]; }
      *(v4f*)(rp + c * 4) = e;
    }
#pragma unroll
    for (int off = 16; off >= 1; off >>= 1) lsum += __shfl_xor(lsum, off, 32);
    if (lane == 0) atomicAdd(&rowsum[row], lsum);
  }
  __syncthreads();
  // Phase 3: normalize in-range, zero-fill masked columns (float4)
  for (int row = 0; row < 64; ++row) {
    float inv = 1.0f / rowsum[row];
    float* rp = attB + (size_t)(q0 + row) * TT;
    for (int c = tid; c < (TT >> 2); c += 256) {
      v4f v;
      if (c * 4 < ncols) {
        v = *(v4f*)(rp + c * 4);
#pragma unroll
        for (int j = 0; j < 4; ++j) v[j] *= inv;
      } else {
        v = v4f{0.f, 0.f, 0.f, 0.f};
      }
      *(v4f*)(rp + c * 4) = v;
    }
  }
}

// ---------------------------------------------------------------------------
// y_head = att @ V. Grid (B*H, T/64). Block computes 64 rows x 64 (=D) cols.
// V is pre-transposed f16 [B,H,D,T] -> B tiles are contiguous: async copies.
// ---------------------------------------------------------------------------
__global__ __launch_bounds__(256)
void mha_av_kernel(const float* __restrict__ att,
                   const _Float16* __restrict__ vT,
                   float* __restrict__ yh)
{
  __shared__ __attribute__((aligned(16))) _Float16 As[64 * 40];
  __shared__ __attribute__((aligned(16))) _Float16 Vt[64 * 40];
  const int tid  = threadIdx.x;
  const int wave = tid >> 5;
  const int bh = blockIdx.x;
  const int qb = blockIdx.y;
  const int q0 = qb * 64;
  const float* attB = att + (size_t)bh * TT * TT;
  const int wm = (wave >> 1) * 16;
  const int wn = (wave & 1) * 32;

  v8f acc[2];
  acc[0] = {}; acc[1] = {};
  const int nsteps = (qb + 1) * 2;   // K up to q0+64, BK=32
  for (int kb = 0; kb < nsteps; ++kb) {
    int k0 = kb * 32;
    __syncthreads();
    // att tile 64x32 fp32 -> f16 (float4 load + b64 LDS store)
    {
      int r  = tid >> 2;               // 0..63
      int c4 = tid & 3;                // chunk of 4 within 32
      v4f f = *(const v4f*)(attB + (size_t)(q0 + r) * TT + k0 + c4 * 4);
      v4h h4;
#pragma unroll
      for (int j = 0; j < 4; ++j) h4[j] = (_Float16)f[j];
      *(v4h*)&As[r * 40 + c4 * 4] = h4;

      v4f f2 = *(const v4f*)(attB + (size_t)(q0 + r) * TT + k0 + (c4 + 4) * 4);
      v4h h42;
#pragma unroll
      for (int j = 0; j < 4; ++j) h42[j] = (_Float16)f2[j];
      *(v4h*)&As[r * 40 + (c4 + 4) * 4] = h42;
    }
    // V tile: rows d=0..63, cols t=k0..k0+31 -> 256 x 8-half async copies
    {
      int d  = tid >> 2;               // 0..63
      int c8 = tid & 3;                // 8-half chunk within 32
      async_copy_b128(&Vt[d * 40 + c8 * 8],
                      vT + ((size_t)bh * DD + d) * TT + k0 + c8 * 8);
    }
    wait_async0();
    __syncthreads();
    v16h a  = frag_ld(&As[wm * 40], 40);
    v16h b0 = frag_ld(&Vt[wn * 40], 40);
    v16h b1 = frag_ld(&Vt[(wn + 16) * 40], 40);
    acc[0] = WMMA_F16(a, b0, acc[0]);
    acc[1] = WMMA_F16(a, b1, acc[1]);
  }

  const int lane = tid & 31;
  const int nl = lane & 15, g = lane >> 4;
#pragma unroll
  for (int sn = 0; sn < 2; ++sn)
#pragma unroll
    for (int r = 0; r < 8; ++r) {
      int row = q0 + wm + r + g * 8;
      int col = wn + sn * 16 + nl;
      yh[((size_t)bh * TT + row) * DD + col] = acc[sn][r];
    }
}

// ---------------------------------------------------------------------------
extern "C" void kernel_launch(void* const* d_in, const int* in_sizes, int n_in,
                              void* d_out, int out_size, void* d_ws, size_t ws_size,
                              hipStream_t stream) {
  const float* q_x = (const float*)d_in[0];
  const float* k_x = (const float*)d_in[1];
  const float* v_x = (const float*)d_in[2];
  const float* Wq  = (const float*)d_in[3];
  const float* bq  = (const float*)d_in[4];
  const float* Wk  = (const float*)d_in[5];
  const float* bk  = (const float*)d_in[6];
  const float* Wv  = (const float*)d_in[7];
  const float* bv  = (const float*)d_in[8];
  const float* Wo  = (const float*)d_in[9];
  const float* bo  = (const float*)d_in[10];

  float* y_out = (float*)d_out;                               // [B,T,C]
  float* att   = y_out + (size_t)BB * TT * CC;                // [B,H,T,T]

  // Workspace: q/k f16 [B,H,T,D], v f16 [B,H,D,T] (8MB each), y_head fp32 (16MB)
  _Float16* qh = (_Float16*)d_ws;
  _Float16* kh = qh + (size_t)MROWS * CC;
  _Float16* vT = kh + (size_t)MROWS * CC;
  float*    yh = (float*)(vT + (size_t)MROWS * CC);

  dim3 gg(CC / 128, MROWS / 64);   // (8, 64)
  mha_gemm_kernel<0, 0><<<gg, 256, 0, stream>>>(q_x, Wq, bq, qh, nullptr);
  mha_gemm_kernel<0, 0><<<gg, 256, 0, stream>>>(k_x, Wk, bk, kh, nullptr);
  mha_gemm_kernel<0, 2><<<gg, 256, 0, stream>>>(v_x, Wv, bv, vT, nullptr);

  dim3 ga(BB * HH, TT / 64);       // (32, 32)
  mha_scores_kernel<<<ga, 256, 0, stream>>>(qh, kh, att);
  mha_av_kernel<<<ga, 256, 0, stream>>>(att, vT, yh);

  mha_gemm_kernel<1, 1><<<gg, 256, 0, stream>>>(yh, Wo, bo, nullptr, y_out);
}
